// BoundaryAwareSmoothAttention_43705587204303
// MI455X (gfx1250) — matrix operand order repaired and checked by reading, not compiled, not run on hardware
//
#include <hip/hip_runtime.h>

// ---------------------------------------------------------------------------
// BoundaryAwareSmoothAttention, fused flash-style implementation for gfx1250.
// B=4, C=64, H=W=64, HW=4096, CH=32.
// Round 4: fix tensor_load_to_lds arity (6-arg clang-23 form); otherwise
// identical to round 3: 128-key blocks, TDM-fed double-buffered K/V tiles,
// DPP16 reductions, exp2-domain softmax, coalesced LDS-transposed epilogue.
// ---------------------------------------------------------------------------

#define Bn 4
#define Cc 64
#define CH 32
#define Hh 64
#define Ww 64
#define HW 4096
#define BN_EPS 1e-5f
#define LOG2E 1.44269504088896340736f

typedef __attribute__((ext_vector_type(16))) _Float16 v16h;
typedef __attribute__((ext_vector_type(8)))  _Float16 v8h;
typedef __attribute__((ext_vector_type(8)))  float    v8f;
typedef __attribute__((ext_vector_type(4)))  unsigned int v4u;
typedef __attribute__((ext_vector_type(8)))  int      v8i;
typedef __attribute__((ext_vector_type(4)))  int      v4i;

// --- DPP16 butterfly reduction over the 16-lane half-row -------------------
template <int CTRL>
__device__ __forceinline__ float dpp_max_step(float x) {
    int y = __builtin_amdgcn_update_dpp(__float_as_int(x), __float_as_int(x),
                                        CTRL, 0xF, 0xF, true);
    return fmaxf(x, __int_as_float(y));
}
template <int CTRL>
__device__ __forceinline__ float dpp_add_step(float x) {
    int y = __builtin_amdgcn_update_dpp(__float_as_int(x), __float_as_int(x),
                                        CTRL, 0xF, 0xF, true);
    return x + __int_as_float(y);
}
__device__ __forceinline__ float red16_max(float x) {
    x = dpp_max_step<0xB1>(x);
    x = dpp_max_step<0x4E>(x);
    x = dpp_max_step<0x141>(x);
    x = dpp_max_step<0x140>(x);
    return x;
}
__device__ __forceinline__ float red16_add(float x) {
    x = dpp_add_step<0xB1>(x);
    x = dpp_add_step<0x4E>(x);
    x = dpp_add_step<0x141>(x);
    x = dpp_add_step<0x140>(x);
    return x;
}

// --- Tensor Data Mover: 2D tile global -> LDS (D# per CDNA5 ISA ch.8) ------
// rows x rowUnits tile of 8-byte elements; global row stride strideUnits.
// LDS receives rows packed back-to-back. One instruction, TENSORcnt-tracked.
__device__ __forceinline__ void tdm_load_2d(unsigned ldsOff, const void* gsrc,
                                            unsigned rowUnits, unsigned rows,
                                            unsigned strideUnits)
{
    unsigned long long ga = (unsigned long long)gsrc;
    v4u g0;
    g0.x = 1u;                                        // count=1 (valid user D#)
    g0.y = ldsOff;                                    // lds_addr
    g0.z = (unsigned)ga;                              // global_addr[31:0]
    g0.w = (unsigned)(ga >> 32) | 0x80000000u;        // ga[56:32] | type=2
    v8i g1;
    g1[0] = 0x00030000;                               // data_size=3 (8B)
    g1[1] = (int)((strideUnits & 0xFFFFu) << 16);     // tensor_dim0[15:0]
    g1[2] = (int)((strideUnits >> 16) | ((rows & 0xFFFFu) << 16)); // td0 hi | td1 lo
    g1[3] = (int)((rows >> 16) | (rowUnits << 16));   // td1 hi | tile_dim0
    g1[4] = (int)rows;                                // tile_dim1 | tile_dim2=0
    g1[5] = (int)strideUnits;                         // tensor_dim0_stride lo32
    g1[6] = 0;                                        // stride0 hi | stride1 lo
    g1[7] = 0;
    v4i zz = {0, 0, 0, 0};
    v8i z8i = {0, 0, 0, 0, 0, 0, 0, 0};
    __builtin_amdgcn_tensor_load_to_lds(g0, g1, zz, zz, z8i, 0);
}

// ---------------------------------------------------------------------------
// Kernel 1: edge gate  w[b,m] = 1 + beta * sigmoid(conv1x1(relu(bn(conv3x3(x)))))
// ---------------------------------------------------------------------------
__global__ __launch_bounds__(256) void edge_w_kernel(
    const float* __restrict__ x, const float* __restrict__ We1,
    const float* __restrict__ be1, const float* __restrict__ bn_w,
    const float* __restrict__ bn_b, const float* __restrict__ bn_mean,
    const float* __restrict__ bn_var, const float* __restrict__ We2,
    const float* __restrict__ be2, const float* __restrict__ beta,
    float* __restrict__ w_out)
{
    int id = blockIdx.x * blockDim.x + threadIdx.x;   // 0..16383
    int b  = id >> 12;
    int n  = id & (HW - 1);
    int y  = n >> 6;
    int xx = n & 63;

    float acc[CH];
#pragma unroll
    for (int h = 0; h < CH; ++h) acc[h] = 0.f;

    for (int dy = -1; dy <= 1; ++dy) {
        int yy = y + dy;
        if (yy < 0 || yy >= Hh) continue;
        for (int dx = -1; dx <= 1; ++dx) {
            int xw = xx + dx;
            if (xw < 0 || xw >= Ww) continue;
            int tap = (dy + 1) * 3 + (dx + 1);
            const float* xp = x + (size_t)b * Cc * HW + yy * Ww + xw;
            for (int i = 0; i < Cc; ++i) {
                float xv = xp[(size_t)i * HW];
#pragma unroll
                for (int h = 0; h < CH; ++h)
                    acc[h] = fmaf(We1[((h * Cc + i) * 9) + tap], xv, acc[h]);
            }
        }
    }

    float s = be2[0];
#pragma unroll
    for (int h = 0; h < CH; ++h) {
        float e = acc[h] + be1[h];
        e = (e - bn_mean[h]) * rsqrtf(bn_var[h] + BN_EPS);
        e = fmaf(e, bn_w[h], bn_b[h]);
        e = fmaxf(e, 0.f);
        s = fmaf(We2[h], e, s);
    }
    float sig = 1.f / (1.f + __expf(-s));
    w_out[id] = fmaf(beta[0], sig, 1.f);
}

// ---------------------------------------------------------------------------
// Kernel 2: Q/K/V 1x1 projections -> f16 (Q pre-scaled by log2 e).
// ---------------------------------------------------------------------------
__global__ __launch_bounds__(256) void qkv_kernel(
    const float* __restrict__ x,
    const float* __restrict__ Wq, const float* __restrict__ bq,
    const float* __restrict__ Wk, const float* __restrict__ bk,
    const float* __restrict__ Wv, const float* __restrict__ bv,
    const float* __restrict__ w_key,
    _Float16* __restrict__ Qh, _Float16* __restrict__ Kh,
    _Float16* __restrict__ Vwh)
{
    __shared__ float xs[Cc][Ww + 1];

    int b  = blockIdx.x >> 6;
    int n0 = (blockIdx.x & 63) << 6;

    for (int idx = threadIdx.x; idx < Cc * 64; idx += blockDim.x) {
        int row = idx >> 6, col = idx & 63;
        xs[row][col] = x[((size_t)b * Cc + row) * HW + n0 + col];
    }
    __syncthreads();

    int p  = threadIdx.x & 63;
    int c0 = threadIdx.x >> 6;          // 0..3
    float wk = w_key[(size_t)b * HW + n0 + p];

    v8h qv0 = {}, qv1 = {}, kv0 = {}, kv1 = {};
#pragma unroll
    for (int k = 0; k < 16; ++k) {
        int c = (c0 << 4) + k;
        float aq = bq[c], ak = bk[c], av = bv[c];
        for (int i = 0; i < Cc; ++i) {
            float xv = xs[i][p];
            aq = fmaf(Wq[c * Cc + i], xv, aq);
            ak = fmaf(Wk[c * Cc + i], xv, ak);
            av = fmaf(Wv[c * Cc + i], xv, av);
        }
        _Float16 qh = (_Float16)(aq * LOG2E);
        _Float16 kh = (_Float16)ak;
        if (k < 8) { qv0[k] = qh; kv0[k] = kh; }
        else       { qv1[k - 8] = qh; kv1[k - 8] = kh; }
        Vwh[((size_t)b * Cc + c) * HW + n0 + p] = (_Float16)(av * wk);
    }
    _Float16* qdst = Qh + ((size_t)b * HW + n0 + p) * Cc + (c0 << 4);
    _Float16* kdst = Kh + ((size_t)b * HW + n0 + p) * Cc + (c0 << 4);
    *(v8h*)qdst = qv0; *(v8h*)(qdst + 8) = qv1;
    *(v8h*)kdst = kv0; *(v8h*)(kdst + 8) = kv1;
}

// ---------------------------------------------------------------------------
// Kernel 3: fused flash attention with edge-weighted L1 renorm.
// 64 queries/WG (4 waves x 16 rows); 128-key blocks; K/V tiles moved by the
// TDM (wave 0 issues, s_wait_tensorcnt gates, barrier publishes).
// ---------------------------------------------------------------------------
__global__ __launch_bounds__(128) void flash_kernel(
    const _Float16* __restrict__ Qh, const _Float16* __restrict__ Kh,
    const _Float16* __restrict__ Vwh, const float* __restrict__ w_key,
    const float* __restrict__ x, const float* __restrict__ gamma_p,
    float* __restrict__ out)
{
    // [0,65536):     2 stages x (kt 16KB [128key][64c] + vt 16KB [64c][128key])
    // [65536,81920): per-wave P staging (4 x 16 x 128 f16)
    // [81920,82944): |w| per key, double buffered (2 x 128 f32)
    // epilogue reuses [0,16640) as float ot[64][65]
    __shared__ __align__(32) unsigned char smem[82944];

    int b    = blockIdx.x >> 6;
    int qb   = (blockIdx.x & 63) << 6;
    int tid  = threadIdx.x;
    int wave = tid >> 5;
    int lane = tid & 31;
    int l16  = lane & 15;
    int half = lane >> 4;
    int m0   = qb + (wave << 4);

    const unsigned smemBase = (unsigned)(unsigned long long)(const void*)smem;
    float* aws = (float*)(smem + 81920);

    // --- Q A-fragments (16 rows x 64 channels), registers -------------------
    v16h qa[2];
    {
        const _Float16* qbase = Qh + ((size_t)b * HW + m0 + l16) * Cc;
#pragma unroll
        for (int kc = 0; kc < 2; ++kc) {
            const _Float16* pp = qbase + kc * 32 + half * 16;
            v8h lo = *(const v8h*)pp;
            v8h hi = *(const v8h*)(pp + 8);
            qa[kc] = __builtin_shufflevector(lo, hi, 0, 1, 2, 3, 4, 5, 6, 7,
                                             8, 9, 10, 11, 12, 13, 14, 15);
        }
    }

    const v8f z8 = {0.f, 0.f, 0.f, 0.f, 0.f, 0.f, 0.f, 0.f};
    v8f o[4];
#pragma unroll
    for (int t = 0; t < 4; ++t) o[t] = z8;

    float runmax[8], dpart[8];
#pragma unroll
    for (int r = 0; r < 8; ++r) { runmax[r] = -3.0e38f; dpart[r] = 0.f; }

    const char* kTilesB = (const char*)(Kh + ((size_t)b * HW) * Cc);
    const char* vBaseB  = (const char*)(Vwh + (size_t)b * Cc * HW);

    // TDM issue of one 128-key stage: K tile = contiguous 16KB row-run,
    // V tile = 64 rows x 256B with 8192B global stride, packed into LDS.
    auto issue_tiles = [&](int stage, int kb) {
        unsigned ko = smemBase + stage * 32768;
        tdm_load_2d(ko, kTilesB + (size_t)kb * (Cc * 2),
                    /*rowUnits=*/2048, /*rows=*/1, /*strideUnits=*/2048);
        tdm_load_2d(ko + 16384, vBaseB + (size_t)kb * 2,
                    /*rowUnits=*/32, /*rows=*/Cc, /*strideUnits=*/HW / 4);
    };

    if (wave == 0) issue_tiles(0, 0);

    const int nb = HW / 128;
    for (int i = 0; i < nb; ++i) {
        int kb    = i << 7;
        int stage = i & 1;
        if (wave == 0 && i + 1 < nb) issue_tiles(stage ^ 1, kb + 128);
        float* awsS = aws + stage * 128;
        awsS[tid] = fabsf(w_key[(size_t)b * HW + kb + tid]);
        if (wave == 0) {
            if (i + 1 < nb) __builtin_amdgcn_s_wait_tensorcnt(2);
            else            __builtin_amdgcn_s_wait_tensorcnt(0);
        }
        __syncthreads();   // current tiles + |w| visible to all waves

        const _Float16 (*kt)[Cc]  = (const _Float16(*)[Cc])(smem + stage * 32768);
        const _Float16 (*vt)[128] = (const _Float16(*)[128])(smem + stage * 32768 + 16384);
        _Float16* pl = (_Float16*)(smem + 65536) + wave * 2048;  // [16][128]

        // --- S = Q (16x64) x K_blk^T (64x128): 16 WMMAs ---------------------
        v8f s[8];
#pragma unroll
        for (int t = 0; t < 8; ++t) s[t] = z8;
#pragma unroll
        for (int t = 0; t < 8; ++t) {
#pragma unroll
            for (int kc = 0; kc < 2; ++kc) {
                v16h bf = *(const v16h*)&kt[t * 16 + l16][kc * 32 + half * 16];
                s[t] = __builtin_amdgcn_wmma_f32_16x16x32_f16(
                    false, qa[kc], false, bf, (short)0, s[t], false, false);
            }
        }

        float aw[8];
#pragma unroll
        for (int t = 0; t < 8; ++t) aw[t] = awsS[t * 16 + l16];

        // --- online softmax in exp2 domain (row = r + 8*half) ---------------
#pragma unroll
        for (int r = 0; r < 8; ++r) {
            float m = fmaxf(fmaxf(fmaxf(s[0][r], s[1][r]), fmaxf(s[2][r], s[3][r])),
                            fmaxf(fmaxf(s[4][r], s[5][r]), fmaxf(s[6][r], s[7][r])));
            m = red16_max(m);
            float nm = fmaxf(runmax[r], m);
            float f  = exp2f(runmax[r] - nm);
            runmax[r] = nm;
            dpart[r] *= f;
#pragma unroll
            for (int t = 0; t < 8; ++t) {
                float p = exp2f(s[t][r] - nm);
                s[t][r] = p;
                dpart[r] = fmaf(p, aw[t], dpart[r]);
            }
#pragma unroll
            for (int t = 0; t < 4; ++t) o[t][r] *= f;
        }

        // --- stage P as f16, re-fragment as A matrix ------------------------
#pragma unroll
        for (int t = 0; t < 8; ++t)
#pragma unroll
            for (int r = 0; r < 8; ++r)
                pl[(r + half * 8) * 128 + t * 16 + l16] = (_Float16)s[t][r];

        // --- O += P (16x128) x Vw (128x64): 16 WMMAs ------------------------
#pragma unroll
        for (int kc = 0; kc < 4; ++kc) {
            const _Float16* pb = pl + l16 * 128 + kc * 32 + half * 16;
            v8h lo = *(const v8h*)pb;
            v8h hi = *(const v8h*)(pb + 8);
            v16h pa = __builtin_shufflevector(lo, hi, 0, 1, 2, 3, 4, 5, 6, 7,
                                              8, 9, 10, 11, 12, 13, 14, 15);
#pragma unroll
            for (int t2 = 0; t2 < 4; ++t2) {
                v16h bf = *(const v16h*)&vt[t2 * 16 + l16][kc * 32 + half * 16];
                o[t2] = __builtin_amdgcn_wmma_f32_16x16x32_f16(
                    false, pa, false, bf, (short)0, o[t2], false, false);
            }
        }
        __syncthreads();   // stage buffers free for the next TDM overwrite
    }

    // --- epilogue: L1 renorm, LDS transpose, coalesced residual store -------
    float* ot = (float*)smem;                 // [64][65] floats, reuses stages
#pragma unroll
    for (int r = 0; r < 8; ++r) {
        float d   = red16_add(dpart[r]);
        float inv = 1.f / fmaxf(d, 1e-12f);
        int nl = (wave << 4) + r + (half << 3);
#pragma unroll
        for (int t2 = 0; t2 < 4; ++t2)
            ot[(t2 * 16 + l16) * 65 + nl] = o[t2][r] * inv;
    }
    __syncthreads();

    float gamma = gamma_p[0];
    for (int idx = tid; idx < 1024; idx += 128) {
        int c  = idx >> 4;
        int j0 = (idx & 15) << 2;
        size_t gi = ((size_t)b * Cc + c) * HW + qb + j0;
        float4 xv = *(const float4*)(x + gi);
        float4 ov;
        ov.x = fmaf(gamma, ot[c * 65 + j0 + 0], xv.x);
        ov.y = fmaf(gamma, ot[c * 65 + j0 + 1], xv.y);
        ov.z = fmaf(gamma, ot[c * 65 + j0 + 2], xv.z);
        ov.w = fmaf(gamma, ot[c * 65 + j0 + 3], xv.w);
        *(float4*)(out + gi) = ov;
    }
}

// ---------------------------------------------------------------------------
extern "C" void kernel_launch(void* const* d_in, const int* in_sizes, int n_in,
                              void* d_out, int out_size, void* d_ws, size_t ws_size,
                              hipStream_t stream)
{
    (void)in_sizes; (void)n_in; (void)out_size; (void)ws_size;
    const float* x     = (const float*)d_in[0];
    const float* Wq    = (const float*)d_in[1];
    const float* bq    = (const float*)d_in[2];
    const float* Wk    = (const float*)d_in[3];
    const float* bk    = (const float*)d_in[4];
    const float* Wv    = (const float*)d_in[5];
    const float* bv    = (const float*)d_in[6];
    const float* We1   = (const float*)d_in[7];
    const float* be1   = (const float*)d_in[8];
    const float* bn_w  = (const float*)d_in[9];
    const float* bn_b  = (const float*)d_in[10];
    const float* bn_mean = (const float*)d_in[11];
    const float* bn_var  = (const float*)d_in[12];
    const float* We2   = (const float*)d_in[13];
    const float* be2   = (const float*)d_in[14];
    const float* gamma = (const float*)d_in[15];
    const float* beta  = (const float*)d_in[16];
    float* out = (float*)d_out;

    char* ws = (char*)d_ws;
    float*    w_key = (float*)ws;                                  //  64 KB
    _Float16* Qhp   = (_Float16*)(ws + (64 << 10));                //   2 MB
    _Float16* Khp   = (_Float16*)(ws + (64 << 10) + (2 << 20));    //   2 MB
    _Float16* Vwhp  = (_Float16*)(ws + (64 << 10) + (4 << 20));    //   2 MB

    edge_w_kernel<<<64, 256, 0, stream>>>(x, We1, be1, bn_w, bn_b, bn_mean,
                                          bn_var, We2, be2, beta, w_key);
    qkv_kernel<<<256, 256, 0, stream>>>(x, Wq, bq, Wk, bk, Wv, bv, w_key,
                                        Qhp, Khp, Vwhp);
    flash_kernel<<<256, 128, 0, stream>>>(Qhp, Khp, Vwhp, w_key, x, gamma, out);
}
